// TAOBAODurendal_35132832481402
// MI455X (gfx1250) — matrix operand order repaired and checked by reading, not compiled, not run on hardware
//
#include <hip/hip_runtime.h>

#define NU 100000
#define NI 50000
#define CD 128
#define NE 1000000
#define NL 200000
#define TAU 0.05f

typedef __attribute__((ext_vector_type(16))) __bf16 v16bf;
typedef __attribute__((ext_vector_type(8)))  float  v8f;

static __device__ __forceinline__ __bf16 f2bf(float f) {
    union { float f; unsigned u; } x; x.f = f;
    unsigned r = x.u + 0x7FFFu + ((x.u >> 16) & 1u);   // round-to-nearest-even
    unsigned short h = (unsigned short)(r >> 16);
    return __builtin_bit_cast(__bf16, h);
}

// native fp32 atomic add (global_atomic_add_f32), relaxed, device scope
static __device__ __forceinline__ void atomAddF(float* p, float v) {
    __hip_atomic_fetch_add(p, v, __ATOMIC_RELAXED, __HIP_MEMORY_SCOPE_AGENT);
}

// ---------------- segment mean ----------------
__global__ __launch_bounds__(256) void count_edges(const int* __restrict__ dst,
                                                   float* __restrict__ cnt, int E_) {
    int e = blockIdx.x * 256 + threadIdx.x;
    if (e < E_) atomAddF(&cnt[dst[e]], 1.0f);
}

// one wave per edge, 4 channels per lane (coalesced across lanes)
__global__ __launch_bounds__(256) void scatter_add(const float* __restrict__ feat,
                                                   const int* __restrict__ src,
                                                   const int* __restrict__ dst,
                                                   float* __restrict__ agg, int E_) {
    int gid = blockIdx.x * 256 + threadIdx.x;
    int e = gid >> 5, lane = gid & 31;
    if (e >= E_) return;
    const float* fp = feat + (size_t)src[e] * CD;
    float* ap = agg + (size_t)dst[e] * CD;
#pragma unroll
    for (int j = 0; j < 4; ++j) {
        int c = lane + (j << 5);
        atomAddF(&ap[c], fp[c]);
    }
}

__global__ __launch_bounds__(256) void normalize_agg(float* __restrict__ agg,
                                                     const float* __restrict__ cnt, int total) {
    int i = blockIdx.x * 256 + threadIdx.x;
    if (i < total) agg[i] = agg[i] / fmaxf(cnt[i >> 7], 1.0f);
}

// ---------------- weight prep (bf16, transposed so K is contiguous) ----------------
// outT[n*CD + k] = bf16( W[k*CD + n] )
__global__ __launch_bounds__(256) void transpose_w(const float* __restrict__ W,
                                                   __bf16* __restrict__ outT) {
    int t = blockIdx.x * 256 + threadIdx.x;
    int k = t >> 7, n = t & 127;
    outT[n * CD + k] = f2bf(W[k * CD + n]);
}

// P = X @ Y (fold projection into relation weight); outT[n*CD + k] = bf16(P[k][n])
__global__ __launch_bounds__(256) void fold_w(const float* __restrict__ X,
                                              const float* __restrict__ Y,
                                              __bf16* __restrict__ outT) {
    int t = blockIdx.x * 256 + threadIdx.x;
    int k = t >> 7, n = t & 127;
    float s = 0.f;
#pragma unroll 4
    for (int j = 0; j < CD; ++j) s += X[k * CD + j] * Y[j * CD + n];
    outT[n * CD + k] = f2bf(s);
}

// ---------------- fused hetero-conv GEMM:  out = blend(past, A@Wa + B@Wb + bias) ----------
// block = 256 (8 waves). Block computes 16 rows x 128 cols; wave w -> cols [16w,16w+16).
__global__ __launch_bounds__(256)
void gemm_wmma(const float* __restrict__ A,       // [N,CD] aggregated neighbors
               const float* __restrict__ Bself,   // [N,CD] self features
               const __bf16* __restrict__ WaT,    // [CD,CD] bf16 transposed
               const __bf16* __restrict__ WbT,
               const float* __restrict__ bias,    // [CD] or nullptr
               const float* __restrict__ past,    // [N,CD]
               const int* __restrict__ snap,
               float* __restrict__ out, int N) {
    __shared__ __bf16 sA[16 * CD];
    __shared__ __bf16 sB[16 * CD];
    int m0 = blockIdx.x << 4;
    for (int i = threadIdx.x; i < 16 * CD; i += 256) {
        int r = i >> 7, c = i & 127;
        size_t g = (size_t)(m0 + r) * CD + c;
        sA[i] = f2bf(A[g]);
        sB[i] = f2bf(Bself[g]);
    }
    __syncthreads();

    int wv = threadIdx.x >> 5;
    int lane = threadIdx.x & 31;
    int hf = lane >> 4;          // lane half (0/1)
    int mr = lane & 15;          // row within tile (A) / col within tile (B)
    int n = (wv << 4) + mr;      // output column this lane owns for B/C/D

    v8f acc = {};
#pragma unroll
    for (int kk = 0; kk < CD; kk += 32) {
        v16bf a, b;
#pragma unroll
        for (int v = 0; v < 8; ++v) {
            int ka = kk + ((v >> 2) << 4) + (hf << 3) + ((v & 3) << 1); // A 16x32 layout
            a[2 * v]     = sA[mr * CD + ka];
            a[2 * v + 1] = sA[mr * CD + ka + 1];
            int kb = kk + ((v >> 2) << 4) + (hf << 3) + ((v & 3) << 1); // B 32x16 layout
            b[2 * v]     = WaT[n * CD + kb];
            b[2 * v + 1] = WaT[n * CD + kb + 1];
        }
        acc = __builtin_amdgcn_wmma_f32_16x16x32_bf16(false, a, false, b,
                                                      (short)0, acc, false, false);
#pragma unroll
        for (int v = 0; v < 8; ++v) {
            int ka = kk + ((v >> 2) << 4) + (hf << 3) + ((v & 3) << 1);
            a[2 * v]     = sB[mr * CD + ka];
            a[2 * v + 1] = sB[mr * CD + ka + 1];
            int kb = kk + ((v >> 2) << 4) + (hf << 3) + ((v & 3) << 1);
            b[2 * v]     = WbT[n * CD + kb];
            b[2 * v + 1] = WbT[n * CD + kb + 1];
        }
        acc = __builtin_amdgcn_wmma_f32_16x16x32_bf16(false, a, false, b,
                                                      (short)0, acc, false, false);
    }

    bool blend = (*snap != 0);
    float bi = bias ? bias[n] : 0.0f;
#pragma unroll
    for (int r = 0; r < 8; ++r) {
        int m = m0 + r + (hf << 3);          // C/D layout: M = r + 8*half
        size_t idx = (size_t)m * CD + n;
        float v = acc[r] + bi;
        if (blend) v = TAU * past[idx] + (1.0f - TAU) * v;
        out[idx] = v;
    }
}

// ---------------- scoring head ----------------
__global__ void prep_wsum(const float* __restrict__ Wpost, const float* __restrict__ bpost,
                          float* __restrict__ wsum, float* __restrict__ bsum) {
    int c = threadIdx.x;
    if (c < CD) wsum[c] = Wpost[2 * c] + Wpost[2 * c + 1];
    if (c == 0) *bsum = bpost[0] + bpost[1];
}

// one wave per label: gather two rows, weighted Hadamard dot, shuffle reduce
__global__ __launch_bounds__(256) void score(const float* __restrict__ uo,
                                             const float* __restrict__ io,
                                             const int* __restrict__ ls,
                                             const int* __restrict__ ld,
                                             const float* __restrict__ wsum,
                                             const float* __restrict__ bsum,
                                             float* __restrict__ h, int Ln) {
    int gid = blockIdx.x * 256 + threadIdx.x;
    int l = gid >> 5, lane = gid & 31;
    if (l >= Ln) return;
    const float* up = uo + (size_t)ls[l] * CD;
    const float* ip = io + (size_t)ld[l] * CD;
    float s = 0.f;
#pragma unroll
    for (int j = 0; j < 4; ++j) {
        int c = lane + (j << 5);
        s += up[c] * ip[c] * wsum[c];
    }
#pragma unroll
    for (int off = 16; off > 0; off >>= 1) s += __shfl_xor(s, off, 32);
    if (lane == 0) h[l] = s + *bsum;
}

extern "C" void kernel_launch(void* const* d_in, const int* in_sizes, int n_in,
                              void* d_out, int out_size, void* d_ws, size_t ws_size,
                              hipStream_t stream) {
    const float* x_user   = (const float*)d_in[0];
    const float* x_item   = (const float*)d_in[1];
    const float* p1u      = (const float*)d_in[2];
    const float* p1i      = (const float*)d_in[3];
    const float* p2u      = (const float*)d_in[4];
    const float* p2i      = (const float*)d_in[5];
    const float* W1ui_n   = (const float*)d_in[6];
    const float* W1ui_s   = (const float*)d_in[7];
    const float* W1iu_n   = (const float*)d_in[8];
    const float* W1iu_s   = (const float*)d_in[9];
    const float* Wp1      = (const float*)d_in[10];
    const float* bp1      = (const float*)d_in[11];
    // d_in[12..14] = Wa1/ba1/q1 : dead (R=1 semantic attention is identity)
    const float* W2ui_n   = (const float*)d_in[15];
    const float* W2ui_s   = (const float*)d_in[16];
    const float* W2iu_n   = (const float*)d_in[17];
    const float* W2iu_s   = (const float*)d_in[18];
    const float* Wp2      = (const float*)d_in[19];
    const float* bp2      = (const float*)d_in[20];
    // d_in[21..23] = Wa2/ba2/q2 : dead
    const float* Wpost    = (const float*)d_in[24];
    const float* bpost    = (const float*)d_in[25];
    const int* src_ui     = (const int*)d_in[26];
    const int* dst_ui     = (const int*)d_in[27];
    const int* src_iu     = (const int*)d_in[28];
    const int* dst_iu     = (const int*)d_in[29];
    const int* lbl_src    = (const int*)d_in[30];
    const int* lbl_dst    = (const int*)d_in[31];
    const int* snap       = (const int*)d_in[32];

    // outputs: h[L], cur1_user, cur1_item, cur2_user, cur2_item (flat concat)
    float* out_h = (float*)d_out;
    float* c1u = out_h + NL;
    float* c1i = c1u + (size_t)NU * CD;
    float* c2u = c1i + (size_t)NI * CD;
    float* c2i = c2u + (size_t)NU * CD;

    // workspace layout
    float* aggUI = (float*)d_ws;                 // [NI,CD]
    float* aggIU = aggUI + (size_t)NI * CD;      // [NU,CD]
    float* cntUI = aggIU + (size_t)NU * CD;      // [NI]
    float* cntIU = cntUI + NI;                   // [NU]
    __bf16* Wt   = (__bf16*)(cntIU + NU);        // 8 x [CD,CD] bf16 transposed
    float* wsum  = (float*)(Wt + 8 * CD * CD);   // [CD]
    float* bsum  = wsum + CD;

    const int EB = (NE + 255) / 256;             // count grid
    const int ESB = (NE * 32 + 255) / 256;       // scatter grid (wave/edge)

    // zero aggs + counts (contiguous)
    hipMemsetAsync(d_ws, 0, (size_t)(NI * CD + NU * CD + NI + NU) * sizeof(float), stream);

    // degree counts (shared by both layers)
    count_edges<<<EB, 256, 0, stream>>>(dst_ui, cntUI, NE);
    count_edges<<<EB, 256, 0, stream>>>(dst_iu, cntIU, NE);

    // ---- layer 1 aggregation ----
    scatter_add<<<ESB, 256, 0, stream>>>(x_user, src_ui, dst_ui, aggUI, NE);
    scatter_add<<<ESB, 256, 0, stream>>>(x_item, src_iu, dst_iu, aggIU, NE);
    normalize_agg<<<(NI * CD + 255) / 256, 256, 0, stream>>>(aggUI, cntUI, NI * CD);
    normalize_agg<<<(NU * CD + 255) / 256, 256, 0, stream>>>(aggIU, cntIU, NU * CD);

    // ---- weight prep: transpose/convert + fold Wp into user relation weights ----
    transpose_w<<<64, 256, 0, stream>>>(W1ui_n, Wt + 0 * CD * CD);
    transpose_w<<<64, 256, 0, stream>>>(W1ui_s, Wt + 1 * CD * CD);
    fold_w<<<64, 256, 0, stream>>>(W1iu_n, Wp1, Wt + 2 * CD * CD);
    fold_w<<<64, 256, 0, stream>>>(W1iu_s, Wp1, Wt + 3 * CD * CD);
    transpose_w<<<64, 256, 0, stream>>>(W2ui_n, Wt + 4 * CD * CD);
    transpose_w<<<64, 256, 0, stream>>>(W2ui_s, Wt + 5 * CD * CD);
    fold_w<<<64, 256, 0, stream>>>(W2iu_n, Wp2, Wt + 6 * CD * CD);
    fold_w<<<64, 256, 0, stream>>>(W2iu_s, Wp2, Wt + 7 * CD * CD);

    // ---- layer 1 GEMMs (NI, NU divisible by 16) ----
    gemm_wmma<<<NI / 16, 256, 0, stream>>>(aggUI, x_item, Wt + 0 * CD * CD, Wt + 1 * CD * CD,
                                           nullptr, p1i, snap, c1i, NI);
    gemm_wmma<<<NU / 16, 256, 0, stream>>>(aggIU, x_user, Wt + 2 * CD * CD, Wt + 3 * CD * CD,
                                           bp1, p1u, snap, c1u, NU);

    // ---- layer 2 aggregation (reuse agg buffers + counts) ----
    hipMemsetAsync(d_ws, 0, (size_t)(NI * CD + NU * CD) * sizeof(float), stream);
    scatter_add<<<ESB, 256, 0, stream>>>(c1u, src_ui, dst_ui, aggUI, NE);
    scatter_add<<<ESB, 256, 0, stream>>>(c1i, src_iu, dst_iu, aggIU, NE);
    normalize_agg<<<(NI * CD + 255) / 256, 256, 0, stream>>>(aggUI, cntUI, NI * CD);
    normalize_agg<<<(NU * CD + 255) / 256, 256, 0, stream>>>(aggIU, cntIU, NU * CD);

    // ---- layer 2 GEMMs ----
    gemm_wmma<<<NI / 16, 256, 0, stream>>>(aggUI, c1i, Wt + 4 * CD * CD, Wt + 5 * CD * CD,
                                           nullptr, p2i, snap, c2i, NI);
    gemm_wmma<<<NU / 16, 256, 0, stream>>>(aggIU, c1u, Wt + 6 * CD * CD, Wt + 7 * CD * CD,
                                           bp2, p2u, snap, c2u, NU);

    // ---- scoring head ----
    prep_wsum<<<1, 128, 0, stream>>>(Wpost, bpost, wsum, bsum);
    score<<<(NL * 32 + 255) / 256, 256, 0, stream>>>(c2u, c2i, lbl_src, lbl_dst,
                                                     wsum, bsum, out_h, NL);
}
